// DecoderLayer_70755291234841
// MI455X (gfx1250) — matrix operand order repaired and checked by reading, not compiled, not run on hardware
//
#include <hip/hip_runtime.h>
#include <hip/hip_bf16.h>

// ---------------------------------------------------------------------------
// MI455X (gfx1250) decoder layer: bf16 WMMA GEMMs + WMMA flash-attention.
// B=2, S=2048, D=1024, H=16, HD=64, FF=4096.  Compute-bound (~206 GFLOP vs
// ~150MB traffic) -> dense math via v_wmma_f32_16x16x32_bf16; LDS tiles are
// staged with CDNA5 async copies (global_load_async_to_lds_b128, ASYNCcnt)
// in a double-buffered pipeline so staging overlaps the WMMA stream.
// ---------------------------------------------------------------------------

typedef __attribute__((ext_vector_type(16))) __bf16    v16bf;
typedef __attribute__((ext_vector_type(8)))  unsigned int v8u;
typedef __attribute__((ext_vector_type(8)))  float     v8f;

static __device__ __forceinline__ unsigned short f2bf(float f) {
  unsigned int u = __float_as_uint(f);
  u += 0x7FFFu + ((u >> 16) & 1u);          // round-to-nearest-even
  return (unsigned short)(u >> 16);
}

static __device__ __forceinline__ v8u pack8(uint4 a, uint4 b) {
  v8u r;
  r[0] = a.x; r[1] = a.y; r[2] = a.z; r[3] = a.w;
  r[4] = b.x; r[5] = b.y; r[6] = b.z; r[7] = b.w;
  return r;
}

static __device__ __forceinline__ v8f wmma_bf16(v8u a, v8u b, v8f c) {
  return __builtin_amdgcn_wmma_f32_16x16x32_bf16(
      /*neg_a=*/false, __builtin_bit_cast(v16bf, a),
      /*neg_b=*/false, __builtin_bit_cast(v16bf, b),
      /*c_mod=*/(short)0, c, /*reuse_a=*/false, /*reuse_b=*/false);
}

// CDNA5 async copy: per-lane 16B global -> LDS, tracked by ASYNCcnt.
// Low 32 bits of a generic shared pointer are the raw LDS byte address
static __device__ __forceinline__ void async_load_b128(const void* gptr,
                                                       void* lptr) {
  unsigned lds = (unsigned)(size_t)lptr;
  unsigned long long ga = (unsigned long long)(size_t)gptr;
  asm volatile("global_load_async_to_lds_b128 %0, %1, off"
               :: "v"(lds), "v"(ga) : "memory");
}
static __device__ __forceinline__ void wait_asynccnt0() {
  asm volatile("s_wait_asynccnt 0" ::: "memory");
}

union U8x16 { uint4 v; unsigned short e[8]; };

// ---------------------------------------------------------------------------
// f32 -> bf16 conversion (vectorized, n multiple of 4)
// ---------------------------------------------------------------------------
__global__ void k_cvt_bf16(const float* __restrict__ s,
                           unsigned short* __restrict__ d, int n4) {
  int i = blockIdx.x * blockDim.x + threadIdx.x;
  if (i >= n4) return;
  float4 v = ((const float4*)s)[i];
  unsigned int lo = (unsigned)f2bf(v.x) | ((unsigned)f2bf(v.y) << 16);
  unsigned int hi = (unsigned)f2bf(v.z) | ((unsigned)f2bf(v.w) << 16);
  ((uint2*)d)[i] = make_uint2(lo, hi);
}

// ---------------------------------------------------------------------------
// Fused downcast + transpose: src f32 [K,N] -> dst bf16 [N,K].
// Done once per weight so GEMM B-tiles become contiguous async copies.
// ---------------------------------------------------------------------------
__global__ __launch_bounds__(256)
void k_cvt_t(const float* __restrict__ s, unsigned short* __restrict__ d,
             int K, int N) {
  __shared__ unsigned short t[32][33];
  const int bx = blockIdx.x * 32;            // N tile
  const int by = blockIdx.y * 32;            // K tile
  const int tx = threadIdx.x & 31, ty = threadIdx.x >> 5;
#pragma unroll
  for (int i = 0; i < 32; i += 8)
    t[ty + i][tx] = f2bf(s[(size_t)(by + ty + i) * N + bx + tx]);
  __syncthreads();
#pragma unroll
  for (int i = 0; i < 32; i += 8)
    d[(size_t)(bx + ty + i) * K + by + tx] = t[tx][ty + i];
}

// ---------------------------------------------------------------------------
// bf16 GEMM:  C[M,N] = A[M,K] * Wt[N,K]^T + bias, optional ReLU,
// optional f32 and/or bf16 outputs.  Tile 128x128, BK=32, 8 waves (2x4),
// wave patch 64x32 = 4x2 wmma tiles.  Double-buffered async LDS staging.
// ---------------------------------------------------------------------------
__global__ __launch_bounds__(256)
void k_gemm_bf16(const unsigned short* __restrict__ A,    // [M,K]
                 const unsigned short* __restrict__ Wt,   // [N,K]
                 const float* __restrict__ bias,
                 float* __restrict__ Cf, unsigned short* __restrict__ Cb,
                 int M, int N, int K, int relu) {
  __shared__ unsigned short As[2][128][40];   // stride 80B, 16B-aligned rows
  __shared__ unsigned short Bs[2][128][40];

  const int tid  = threadIdx.x;
  const int lane = tid & 31, wave = tid >> 5;
  const int half = lane >> 4, l15 = lane & 15;
  const int wr = wave >> 2, wc = wave & 3;         // 2x4 wave grid
  const int m0 = blockIdx.y * 128, n0 = blockIdx.x * 128;
  const int nk = K >> 5;

  v8f zero8 = {0.f,0.f,0.f,0.f,0.f,0.f,0.f,0.f};
  v8f acc[4][2];
#pragma unroll
  for (int i = 0; i < 4; ++i)
#pragma unroll
    for (int j = 0; j < 2; ++j) acc[i][j] = zero8;

  auto stage = [&](int kt, int buf) {
    const int k0 = kt * 32;
#pragma unroll
    for (int it = 0; it < 2; ++it) {
      int idx = tid + it * 256;                    // 0..511
      int row = idx >> 2, c = idx & 3;             // 4 chunks of 8 bf16
      async_load_b128(A  + (size_t)(m0 + row) * K + k0 + c * 8,
                      &As[buf][row][c * 8]);
      async_load_b128(Wt + (size_t)(n0 + row) * K + k0 + c * 8,
                      &Bs[buf][row][c * 8]);
    }
  };

  stage(0, 0);
  wait_asynccnt0();
  __syncthreads();

  for (int kt = 0; kt < nk; ++kt) {
    const int cur = kt & 1;
    if (kt + 1 < nk) stage(kt + 1, cur ^ 1);      // overlaps with WMMAs below

    v8u af[4], bfrag[2];
#pragma unroll
    for (int mt = 0; mt < 4; ++mt) {
      const unsigned short* p = &As[cur][wr * 64 + mt * 16 + l15][0];
      af[mt] = pack8(*(const uint4*)(p + 8 * half),
                     *(const uint4*)(p + 16 + 8 * half));
    }
#pragma unroll
    for (int nt = 0; nt < 2; ++nt) {
      const unsigned short* p = &Bs[cur][wc * 32 + nt * 16 + l15][0];
      bfrag[nt] = pack8(*(const uint4*)(p + 8 * half),
                        *(const uint4*)(p + 16 + 8 * half));
    }
#pragma unroll
    for (int mt = 0; mt < 4; ++mt)
#pragma unroll
      for (int nt = 0; nt < 2; ++nt)
        acc[mt][nt] = wmma_bf16(af[mt], bfrag[nt], acc[mt][nt]);

    wait_asynccnt0();
    __syncthreads();
  }

  // ---- epilogue: bias (+ReLU), f32 and/or bf16 stores ---------------------
#pragma unroll
  for (int mt = 0; mt < 4; ++mt) {
#pragma unroll
    for (int nt = 0; nt < 2; ++nt) {
      int col = n0 + wc * 32 + nt * 16 + l15;
      float bv = bias[col];
#pragma unroll
      for (int j = 0; j < 8; ++j) {
        int row = m0 + wr * 64 + mt * 16 + j + 8 * half;
        float v = acc[mt][nt][j] + bv;
        if (relu) v = v > 0.f ? v : 0.f;
        size_t o = (size_t)row * N + col;
        if (Cf) Cf[o] = v;
        if (Cb) Cb[o] = f2bf(v);
      }
    }
  }
}

// ---------------------------------------------------------------------------
// Flash attention (HD=64).  Grid: (S/128, H, B), 8 waves / WG.
// Wave owns 16 query rows; keys streamed in chunks of 64 through LDS (K tile
// staged with async copies).  QK^T and P*V via bf16 WMMA.
// ---------------------------------------------------------------------------
__global__ __launch_bounds__(256)
void k_flash(const unsigned short* __restrict__ qb, int qs,
             const unsigned short* __restrict__ kb, int ks, int kc0,
             const unsigned short* __restrict__ vb, int vs, int vc0,
             unsigned short* __restrict__ ob, int os,
             int S, int causal) {
  __shared__ unsigned short Kl[64][72];        // [key][hd]
  __shared__ unsigned short Vt[64][72];        // [hd][key] (transposed)
  __shared__ unsigned short Pl[8][16][72];     // per-wave probs, bf16

  const int tid  = threadIdx.x;
  const int lane = tid & 31, wave = tid >> 5;
  const int half = lane >> 4, l15 = lane & 15;
  const int tile = blockIdx.x, h = blockIdx.y, b = blockIdx.z;
  const size_t rb = (size_t)b * S;

  const unsigned short* Q  = qb + rb * qs + h * 64;
  const unsigned short* Kp = kb + rb * ks + kc0 + h * 64;
  const unsigned short* Vp = vb + rb * vs + vc0 + h * 64;
  unsigned short*       Op = ob + rb * os + h * 64;

  const int qbase = tile * 128 + wave * 16;

  // Q fragments (16 rows x 64 hd = 2 A-operands), loaded once
  v8u qf[2];
#pragma unroll
  for (int kst = 0; kst < 2; ++kst) {
    const unsigned short* p = Q + (size_t)(qbase + l15) * qs + kst * 32;
    qf[kst] = pack8(*(const uint4*)(p + 8 * half),
                    *(const uint4*)(p + 16 + 8 * half));
  }

  v8f zero8 = {0.f,0.f,0.f,0.f,0.f,0.f,0.f,0.f};
  float rm[8], rl[8];
  v8f oacc[4];
#pragma unroll
  for (int j = 0; j < 8; ++j) { rm[j] = -3.0e38f; rl[j] = 0.f; }
#pragma unroll
  for (int ot = 0; ot < 4; ++ot) oacc[ot] = zero8;

  const int kend = causal ? (tile * 128 + 128) : S;
  for (int kc = 0; kc < kend; kc += 64) {
    // ---- stage K chunk (async -> LDS) + V chunk (register transpose) ----
#pragma unroll
    for (int it = 0; it < 2; ++it) {
      int idx = tid + it * 256;                 // 0..511
      int row = idx >> 3, c = idx & 7;          // 8 chunks of 8 per row
      async_load_b128(Kp + (size_t)(kc + row) * ks + c * 8, &Kl[row][c * 8]);
      U8x16 u;
      u.v = *(const uint4*)(Vp + (size_t)(kc + row) * vs + c * 8);
#pragma unroll
      for (int i = 0; i < 8; ++i) Vt[c * 8 + i][row] = u.e[i];
    }
    wait_asynccnt0();
    __syncthreads();

    // ---- scores = Q * K^T  (4 ntiles x 2 ksteps of WMMA) ----------------
    v8f st[4];
#pragma unroll
    for (int nt = 0; nt < 4; ++nt) {
      v8f s = zero8;
#pragma unroll
      for (int kst = 0; kst < 2; ++kst) {
        const unsigned short* p = &Kl[nt * 16 + l15][kst * 32 + 8 * half];
        v8u kf = pack8(*(const uint4*)p, *(const uint4*)(p + 16));
        s = wmma_bf16(qf[kst], kf, s);
      }
      st[nt] = s;
    }

    // ---- streaming softmax (C-layout: row = j+8*half, col = l15) --------
#pragma unroll
    for (int j = 0; j < 8; ++j) {
      int qr = qbase + j + 8 * half;
      float sc[4];
      float m = -3.0e38f;
#pragma unroll
      for (int nt = 0; nt < 4; ++nt) {
        float x = st[nt][j] * 0.125f;          // 1/sqrt(64)
        int key = kc + nt * 16 + l15;
        if (causal && key > qr) x = -3.0e38f;
        sc[nt] = x;
        m = fmaxf(m, x);
      }
#pragma unroll
      for (int off = 1; off < 16; off <<= 1)
        m = fmaxf(m, __shfl_xor(m, off, 32));  // butterfly within half-wave
      float nm = fmaxf(rm[j], m);
      float alpha = (rm[j] > -1.0e38f) ? __expf(rm[j] - nm) : 0.f;
      float rs = 0.f;
#pragma unroll
      for (int nt = 0; nt < 4; ++nt) {
        float p = (sc[nt] > -1.0e38f) ? __expf(sc[nt] - nm) : 0.f;
        rs += p;
        Pl[wave][j + 8 * half][nt * 16 + l15] = f2bf(p);
      }
#pragma unroll
      for (int off = 1; off < 16; off <<= 1) rs += __shfl_xor(rs, off, 32);
      rl[j] = rl[j] * alpha + rs;
      rm[j] = nm;
#pragma unroll
      for (int ot = 0; ot < 4; ++ot) oacc[ot][j] = oacc[ot][j] * alpha;
    }

    // ---- O += P * V  (reload P in A-fragment layout from wave-local LDS)
    v8u pf[2];
#pragma unroll
    for (int kst = 0; kst < 2; ++kst) {
      const unsigned short* p = &Pl[wave][l15][kst * 32];
      pf[kst] = pack8(*(const uint4*)(p + 8 * half),
                      *(const uint4*)(p + 16 + 8 * half));
    }
#pragma unroll
    for (int ot = 0; ot < 4; ++ot) {
#pragma unroll
      for (int kst = 0; kst < 2; ++kst) {
        const unsigned short* p = &Vt[ot * 16 + l15][kst * 32 + 8 * half];
        v8u vf = pack8(*(const uint4*)p, *(const uint4*)(p + 16));
        oacc[ot] = wmma_bf16(pf[kst], vf, oacc[ot]);
      }
    }
    __syncthreads();
  }

  // ---- normalize + store bf16 ---------------------------------------------
#pragma unroll
  for (int ot = 0; ot < 4; ++ot) {
#pragma unroll
    for (int j = 0; j < 8; ++j) {
      int row = qbase + j + 8 * half;
      float inv = (rl[j] > 0.f) ? 1.0f / rl[j] : 0.f;
      Op[(size_t)row * os + ot * 16 + l15] = f2bf(oacc[ot][j] * inv);
    }
  }
}

// ---------------------------------------------------------------------------
// out = LayerNorm(a + b) * g + beta   (one 256-thread block per row, D=1024)
// Writes f32 and optionally a bf16 copy.
// ---------------------------------------------------------------------------
__global__ __launch_bounds__(256)
void k_add_ln(const float* __restrict__ a, const float* __restrict__ b,
              const float* __restrict__ g, const float* __restrict__ beta,
              float* __restrict__ outf, unsigned short* __restrict__ outb) {
  __shared__ float red[256];
  const int row = blockIdx.x, t = threadIdx.x;
  const float4 va = ((const float4*)(a + (size_t)row * 1024))[t];
  const float4 vb = ((const float4*)(b + (size_t)row * 1024))[t];
  float4 v = make_float4(va.x + vb.x, va.y + vb.y, va.z + vb.z, va.w + vb.w);

  red[t] = v.x + v.y + v.z + v.w;
  __syncthreads();
  for (int o = 128; o > 0; o >>= 1) {
    if (t < o) red[t] += red[t + o];
    __syncthreads();
  }
  float mean = red[0] * (1.0f / 1024.0f);
  __syncthreads();

  float dx = v.x - mean, dy = v.y - mean, dz = v.z - mean, dw = v.w - mean;
  red[t] = dx * dx + dy * dy + dz * dz + dw * dw;
  __syncthreads();
  for (int o = 128; o > 0; o >>= 1) {
    if (t < o) red[t] += red[t + o];
    __syncthreads();
  }
  float rstd = rsqrtf(red[0] * (1.0f / 1024.0f) + 1e-9f);

  const float4 gg = ((const float4*)g)[t];
  const float4 bb = ((const float4*)beta)[t];
  float4 o4;
  o4.x = gg.x * dx * rstd + bb.x;
  o4.y = gg.y * dy * rstd + bb.y;
  o4.z = gg.z * dz * rstd + bb.z;
  o4.w = gg.w * dw * rstd + bb.w;
  ((float4*)(outf + (size_t)row * 1024))[t] = o4;
  if (outb) {
    unsigned int lo = (unsigned)f2bf(o4.x) | ((unsigned)f2bf(o4.y) << 16);
    unsigned int hi = (unsigned)f2bf(o4.z) | ((unsigned)f2bf(o4.w) << 16);
    ((uint2*)(outb + (size_t)row * 1024))[t] = make_uint2(lo, hi);
  }
}

// ---------------------------------------------------------------------------
// Host orchestration
// ---------------------------------------------------------------------------
extern "C" void kernel_launch(void* const* d_in, const int* in_sizes, int n_in,
                              void* d_out, int out_size, void* d_ws, size_t ws_size,
                              hipStream_t stream) {
  (void)in_sizes; (void)n_in; (void)out_size; (void)ws_size;

  const float* x       = (const float*)d_in[0];
  const float* y       = (const float*)d_in[1];
  // d_in[2] = decoder_mask (tril) -> handled analytically as causal
  const float* qkv_w   = (const float*)d_in[3];
  const float* qkv_b   = (const float*)d_in[4];
  const float* sa_fc_w = (const float*)d_in[5];
  const float* sa_fc_b = (const float*)d_in[6];
  const float* g1      = (const float*)d_in[7];
  const float* b1      = (const float*)d_in[8];
  const float* kv_w    = (const float*)d_in[9];
  const float* kv_b    = (const float*)d_in[10];
  const float* q_w     = (const float*)d_in[11];
  const float* q_b     = (const float*)d_in[12];
  const float* ca_fc_w = (const float*)d_in[13];
  const float* ca_fc_b = (const float*)d_in[14];
  const float* g2      = (const float*)d_in[15];
  const float* b2      = (const float*)d_in[16];
  const float* ff1_w   = (const float*)d_in[17];
  const float* ff1_b   = (const float*)d_in[18];
  const float* ff2_w   = (const float*)d_in[19];
  const float* ff2_b   = (const float*)d_in[20];
  const float* g3      = (const float*)d_in[21];
  const float* b3      = (const float*)d_in[22];

  constexpr int B = 2, S = 2048, D = 1024, H = 16, FF = 4096;
  constexpr int BS = B * S;

  char* ws = (char*)d_ws;
  size_t off = 0;
  auto alloc_bf = [&](size_t elems) -> unsigned short* {
    unsigned short* p = (unsigned short*)(ws + off);
    off += ((elems * 2 + 255) & ~(size_t)255);
    return p;
  };
  auto alloc_f = [&](size_t elems) -> float* {
    float* p = (float*)(ws + off);
    off += ((elems * 4 + 255) & ~(size_t)255);
    return p;
  };

  unsigned short* xbf    = alloc_bf((size_t)BS * D);
  unsigned short* ybf    = alloc_bf((size_t)BS * D);
  unsigned short* wqkv   = alloc_bf((size_t)D * 3 * D);   // transposed [3D, D]
  unsigned short* wsa    = alloc_bf((size_t)D * D);       // transposed [D, D]
  unsigned short* wkv    = alloc_bf((size_t)D * 2 * D);   // transposed [2D, D]
  unsigned short* wq     = alloc_bf((size_t)D * D);
  unsigned short* wca    = alloc_bf((size_t)D * D);
  unsigned short* wf1    = alloc_bf((size_t)D * FF);      // transposed [FF, D]
  unsigned short* wf2    = alloc_bf((size_t)FF * D);      // transposed [D, FF]
  unsigned short* qkvbf  = alloc_bf((size_t)BS * 3 * D);
  unsigned short* attnbf = alloc_bf((size_t)BS * D);
  unsigned short* kvbf   = alloc_bf((size_t)BS * 2 * D);
  unsigned short* q2bf   = alloc_bf((size_t)BS * D);
  unsigned short* cabf   = alloc_bf((size_t)BS * D);
  unsigned short* y1bf   = alloc_bf((size_t)BS * D);
  unsigned short* y2bf   = alloc_bf((size_t)BS * D);
  unsigned short* hbf    = alloc_bf((size_t)BS * FF);
  float* saf = alloc_f((size_t)BS * D);
  float* y1f = alloc_f((size_t)BS * D);
  float* caf = alloc_f((size_t)BS * D);
  float* y2f = alloc_f((size_t)BS * D);
  float* fff = alloc_f((size_t)BS * D);

  auto cvt = [&](const float* s, unsigned short* dst, size_t n) {
    int n4 = (int)(n / 4);
    k_cvt_bf16<<<(n4 + 255) / 256, 256, 0, stream>>>(s, dst, n4);
  };
  auto cvt_t = [&](const float* s, unsigned short* dst, int K, int N) {
    dim3 g(N / 32, K / 32);
    k_cvt_t<<<g, 256, 0, stream>>>(s, dst, K, N);
  };
  auto gemm = [&](const unsigned short* A, const unsigned short* Wt,
                  const float* bias, float* Cf, unsigned short* Cb,
                  int M, int N, int K, int relu) {
    dim3 g(N / 128, M / 128);
    k_gemm_bf16<<<g, 256, 0, stream>>>(A, Wt, bias, Cf, Cb, M, N, K, relu);
  };

  // ---- precision downcast; weights also transposed to [N,K] ----
  cvt(x, xbf, (size_t)BS * D);
  cvt(y, ybf, (size_t)BS * D);
  cvt_t(qkv_w,   wqkv, D, 3 * D);
  cvt_t(sa_fc_w, wsa,  D, D);
  cvt_t(kv_w,    wkv,  D, 2 * D);
  cvt_t(q_w,     wq,   D, D);
  cvt_t(ca_fc_w, wca,  D, D);
  cvt_t(ff1_w,   wf1,  D, FF);
  cvt_t(ff2_w,   wf2,  FF, D);

  // ---- masked self-attention on y ----
  gemm(ybf, wqkv, qkv_b, nullptr, qkvbf, BS, 3 * D, D, 0);
  {
    dim3 g(S / 128, H, B);
    k_flash<<<g, 256, 0, stream>>>(qkvbf, 3 * D,
                                   qkvbf, 3 * D, D,
                                   qkvbf, 3 * D, 2 * D,
                                   attnbf, D, S, /*causal=*/1);
  }
  gemm(attnbf, wsa, sa_fc_b, saf, nullptr, BS, D, D, 0);
  k_add_ln<<<BS, 256, 0, stream>>>(saf, y, g1, b1, y1f, y1bf);

  // ---- cross-attention: kv from x, q from y1 ----
  gemm(xbf, wkv, kv_b, nullptr, kvbf, BS, 2 * D, D, 0);
  gemm(y1bf, wq, q_b, nullptr, q2bf, BS, D, D, 0);
  {
    dim3 g(S / 128, H, B);
    k_flash<<<g, 256, 0, stream>>>(q2bf, D,
                                   kvbf, 2 * D, 0,
                                   kvbf, 2 * D, D,
                                   cabf, D, S, /*causal=*/0);
  }
  gemm(cabf, wca, ca_fc_b, caf, nullptr, BS, D, D, 0);
  k_add_ln<<<BS, 256, 0, stream>>>(caf, y1f, g2, b2, y2f, y2bf);

  // ---- FFN ----
  gemm(y2bf, wf1, ff1_b, nullptr, hbf, BS, FF, D, /*relu=*/1);
  gemm(hbf, wf2, ff2_b, fff, nullptr, BS, D, FF, 0);
  k_add_ln<<<BS, 256, 0, stream>>>(fff, y2f, g3, b3, (float*)d_out, nullptr);
}
